// SkeletalEncBlock_3350074491470
// MI455X (gfx1250) — compile-verified, matrix operand order checked by reading, not run on hardware
//
#include <hip/hip_runtime.h>
#include <hip/hip_bf16.h>
#include <cstddef>

// ---------------------------------------------------------------------------
// Skeletal conv (topology-masked temporal conv1d) + pair pooling + LeakyReLU
// fused into a WMMA f16 GEMM for gfx1250 (MI455X, wave32).
//
// Shapes: x[B=32][768][T=4096] f32, weight[768][768][15] f32 (block tri-diag
// mask), bias[768], out[B][384][T] f32.
//
// Compute: 1.45e11 MAC (290 GFLOP) vs 0.6 GB HBM -> compute bound, so inputs
// are converted to f16 and fed to v_wmma_f32_16x16x32_f16 (f32 accumulate).
// Each wave owns a 32x32 output tile (2 M-tiles x 2 N-tiles) so every
// A-fragment load feeds two WMMAs and every B-fragment feeds two WMMAs.
// ---------------------------------------------------------------------------

typedef __attribute__((ext_vector_type(16))) _Float16 v16h;
typedef __attribute__((ext_vector_type(8)))  _Float16 v8h;
typedef __attribute__((ext_vector_type(8)))  float    v8f;

#define NJ    24      // joints
#define NC    32      // channels per joint (in == out)
#define NCH   768     // NJ*NC
#define KT    15      // temporal taps
#define TPAD  7
#define NT    4096    // frames
#define NB    32      // batch
#define TN    128     // frames per workgroup tile (4 waves x 32 frames)
#define NCOL  142     // TN + 2*TPAD  (x tile columns)
#define CST   136     // cin stride of transposed x tile (128 + 8 pad, 16B mult)

// A-fragment / B-fragment K mapping for 16-bit WMMA 16x16x32 (ISA 7.12.2):
//   k(e, lane) = ((e>>3)<<4) + (e&7) + ((lane>>4)<<3)

// ---------------------------------------------------------------------------
// Prep: swizzle masked weights into per-lane A-fragment order, f16, x0.5
// (the 0.5 implements the pair-average pooling inside the accumulator).
// Layout: wfrag[j][tap][kstep(3)][mtile(2)][lane(32)][e(16)] halves = 2.2 MB.
// ---------------------------------------------------------------------------
__global__ void prep_weights_kernel(const float* __restrict__ weight,
                                    _Float16* __restrict__ wfrag) {
    int idx = blockIdx.x * blockDim.x + threadIdx.x;      // one e-pair each
    const int total = NJ * KT * 3 * 2 * 32 * 8;
    if (idx >= total) return;
    int ep   = idx & 7;
    int lane = (idx >> 3) & 31;
    int mt   = (idx >> 8) & 1;
    int ks   = (idx >> 9) % 3;
    int kj   = (idx >> 9) / 3;      // j*15 + tap
    int tap  = kj % KT;
    int j    = kj / KT;
    int half = lane >> 4;
    int m    = lane & 15;
    int cout = mt * 16 + m;
    #pragma unroll
    for (int t2 = 0; t2 < 2; ++t2) {
        int e     = ep * 2 + t2;
        int kmap  = ((e >> 3) << 4) + (e & 7) + half * 8;   // 0..31
        int cin96 = ks * 32 + kmap;                          // 0..95
        int nj    = j - 1 + (cin96 >> 5);                    // neighbor joint
        float w = 0.0f;
        if (nj >= 0 && nj < NJ) {
            int o = j * NC + cout;
            int i = nj * NC + (cin96 & 31);
            w = 0.5f * weight[((size_t)o * NCH + i) * KT + tap];  // mask + pool
        }
        wfrag[(size_t)idx * 2 + t2] = (_Float16)w;
    }
}

// ---------------------------------------------------------------------------
// Main fused kernel. Grid: (t-tile=32, joint-pair=12, batch=32), block 128.
// ---------------------------------------------------------------------------
__global__ __launch_bounds__(128)
void skel_conv_pool_kernel(const float* __restrict__ x,
                           const _Float16* __restrict__ wfrag,
                           const float* __restrict__ bias,
                           float* __restrict__ out) {
    // x tile transposed to [t][cin]: B-fragments become 2x ds_load_b128/lane.
    __shared__ __align__(16) _Float16 Xt[NCOL * CST];   // 38,624 B

    const int tile = blockIdx.x;          // 0..31
    const int jp   = blockIdx.y;          // 0..11 pooled pair
    const int b    = blockIdx.z;          // 0..31
    const int t0   = tile * TN;
    const int j2   = jp * 2;
    const int tid  = threadIdx.x;

    // ---- stage x tile: 4 joints (j2-1..j2+2) x 128 cin x 142 frames, f32->f16
    const int chbase = (j2 - 1) * NC;     // may be negative at jp==0
    for (int idx = tid; idx < 128 * 144; idx += 128) {
        int col = idx % 144;              // consecutive tid -> consecutive t
        if (col < NCOL) {
            int ci  = idx / 144;
            int gch = chbase + ci;
            int t   = t0 - TPAD + col;
            float v = 0.0f;
            if (gch >= 0 && gch < NCH && t >= 0 && t < NT)
                v = x[((size_t)b * NCH + gch) * NT + t];
            Xt[col * CST + ci] = (_Float16)v;
        }
    }
    __syncthreads();

    const int lane  = tid & 31;
    const int wave  = tid >> 5;           // 0..3
    const int n0    = wave * 32;          // this wave's 32-frame sub-tile
    const int laneN = lane & 15;
    const int kb8   = (lane >> 4) * 8;

    v8f acc00 = {};   // (cout  0..15, frames n0   ..n0+15)
    v8f acc10 = {};   // (cout 16..31, frames n0   ..n0+15)
    v8f acc01 = {};   // (cout  0..15, frames n0+16..n0+31)
    v8f acc11 = {};   // (cout 16..31, frames n0+16..n0+31)

    const _Float16* wbase =
        wfrag + (size_t)j2 * KT * 3 * 2 * 512 + lane * 16;

    for (int k = 0; k < KT; ++k) {
        const _Float16* xcolA = &Xt[(n0 + k + laneN) * CST];
        const _Float16* xcolB = xcolA + 16 * CST;
        #pragma unroll
        for (int h = 0; h < 2; ++h) {           // joint j2+h of the pair
            #pragma unroll
            for (int ks = 0; ks < 3; ++ks) {    // neighbor cin group of 32
                const int cinb = h * 32 + ks * 32 + kb8;
                v8h aL = *(const v8h*)(xcolA + cinb);        // K runs 0..7
                v8h aH = *(const v8h*)(xcolA + cinb + 16);   // K runs 16..23
                v8h bL = *(const v8h*)(xcolB + cinb);
                v8h bH = *(const v8h*)(xcolB + cinb + 16);
                v16h bmA, bmB;
                #pragma unroll
                for (int i = 0; i < 8; ++i) {
                    bmA[i] = aL[i]; bmA[i + 8] = aH[i];
                    bmB[i] = bL[i]; bmB[i + 8] = bH[i];
                }
                const _Float16* wp =
                    wbase + (size_t)(((h * KT + k) * 3 + ks) * 2) * 512;
                v16h a0 = *(const v16h*)(wp);          // mtile 0 (coalesced, L2)
                v16h a1 = *(const v16h*)(wp + 512);    // mtile 1
                acc00 = __builtin_amdgcn_wmma_f32_16x16x32_f16(
                    false, a0, false, bmA, (short)0, acc00, false, false);
                acc01 = __builtin_amdgcn_wmma_f32_16x16x32_f16(
                    false, a0, false, bmB, (short)0, acc01, false, false);
                acc10 = __builtin_amdgcn_wmma_f32_16x16x32_f16(
                    false, a1, false, bmA, (short)0, acc10, false, false);
                acc11 = __builtin_amdgcn_wmma_f32_16x16x32_f16(
                    false, a1, false, bmB, (short)0, acc11, false, false);
            }
        }
    }

    // ---- epilogue: pooled bias + LeakyReLU(0.2), store f32
    // C/D layout: VGPR r -> row kb8+r, col laneN.
    const float* bias0 = bias + j2 * NC;
    float* outp = out + ((size_t)b * (NJ / 2) * NC + jp * NC) * NT
                      + t0 + n0 + laneN;
    #pragma unroll
    for (int r = 0; r < 8; ++r) {
        int m = kb8 + r;
        {
            int co = m;                                   // mtile 0
            float bv = 0.5f * (bias0[co] + bias0[NC + co]);
            float v0 = acc00[r] + bv;
            v0 = (v0 >= 0.0f) ? v0 : 0.2f * v0;
            outp[(size_t)co * NT] = v0;
            float v1 = acc01[r] + bv;
            v1 = (v1 >= 0.0f) ? v1 : 0.2f * v1;
            outp[(size_t)co * NT + 16] = v1;
        }
        {
            int co = 16 + m;                              // mtile 1
            float bv = 0.5f * (bias0[co] + bias0[NC + co]);
            float v0 = acc10[r] + bv;
            v0 = (v0 >= 0.0f) ? v0 : 0.2f * v0;
            outp[(size_t)co * NT] = v0;
            float v1 = acc11[r] + bv;
            v1 = (v1 >= 0.0f) ? v1 : 0.2f * v1;
            outp[(size_t)co * NT + 16] = v1;
        }
    }
}

// ---------------------------------------------------------------------------
extern "C" void kernel_launch(void* const* d_in, const int* in_sizes, int n_in,
                              void* d_out, int out_size, void* d_ws, size_t ws_size,
                              hipStream_t stream) {
    const float* x      = (const float*)d_in[0];
    const float* weight = (const float*)d_in[1];
    const float* bias   = (const float*)d_in[2];
    // d_in[3] (mask) and d_in[4] (pool_pairs) are structural; hardcoded above.
    _Float16* wfrag = (_Float16*)d_ws;   // 2,211,840 bytes of workspace

    const int totalPairs = NJ * KT * 3 * 2 * 32 * 8;   // 552,960
    prep_weights_kernel<<<(totalPairs + 255) / 256, 256, 0, stream>>>(weight, wfrag);

    dim3 grid(NT / TN, NJ / 2, NB);      // (32, 12, 32)
    skel_conv_pool_kernel<<<grid, 128, 0, stream>>>(x, wfrag, bias, (float*)d_out);
}
